// CNN2DImplemented_51994874085714
// MI455X (gfx1250) — matrix operand order repaired and checked by reading, not compiled, not run on hardware
//
#include <hip/hip_runtime.h>
#include <stdint.h>

typedef __attribute__((ext_vector_type(16))) __bf16 v16bf;
typedef __attribute__((ext_vector_type(8)))  float  v8f;

#define CIN    64
#define COUT   128
#define HW     112
#define HWHW   12544            // 112*112
#define NTOT   200704           // 16*HWHW
#define HP     114              // padded height/width
#define HPHP   12996            // 114*114
#define XP_ELEMS (16*CIN*HPHP)  // 13,307,904 packed dwords
#define KCH    18               // 576/32 k-chunks
#define NBLK   128              // spatial positions per block
#define APLANE 4096             // 8 mtiles * 32 lanes * 16 elems per k-chunk
#define A_ELEMS (KCH*APLANE)    // 73728

// ---------------------------------------------------------------------------
// Pre-convert x (NCHW f32) into padded packed bf16 hi/lo:
//   xp[b][c][ih][iw] (114x114, halo = 0) = lo16: hi(bf16), hi16: lo(bf16)
// ---------------------------------------------------------------------------
__global__ __launch_bounds__(256) void prep_x(const float* __restrict__ x,
                                              uint32_t* __restrict__ xp) {
  int p  = blockIdx.x * 256 + threadIdx.x;      // 0..XP_ELEMS-1
  int iw = p % HP;
  int t  = p / HP;
  int ih = t % HP;
  int bc = t / HP;                              // b*64 + c
  uint32_t packed = 0u;
  if (ih >= 1 && ih <= HW && iw >= 1 && iw <= HW) {
    float v  = x[bc * HWHW + (ih - 1) * HW + (iw - 1)];
    __bf16 h = (__bf16)v;
    __bf16 l = (__bf16)(v - (float)h);
    union { __bf16 b; uint16_t u; } uh{h}, ul{l};
    packed = (uint32_t)uh.u | ((uint32_t)ul.u << 16);
  }
  xp[p] = packed;
}

// ---------------------------------------------------------------------------
// Swizzle weights into per-lane WMMA A layout (hi/lo bf16), with the K
// dimension PERMUTED to k_lin = (kh*3+kw)*64 + c  (chunk => uniform kh,kw).
// 16-bit A 16x32 layout (contiguous K): lanes 0-15: K = e ;
//                                       lanes16-31: K = e+16.
// ---------------------------------------------------------------------------
__global__ __launch_bounds__(256) void prep_wswz(const float* __restrict__ w,
                                                 __bf16* __restrict__ Ahi,
                                                 __bf16* __restrict__ Alo) {
  int f    = blockIdx.x * 256 + threadIdx.x;     // 0..A_ELEMS-1
  int e    = f & 15;
  int t    = f >> 4;
  int lane = t & 31;
  int g    = t >> 5;
  int mt   = g & 7;
  int kc   = g >> 3;
  int K    = e + 16 * (lane >> 4);
  int klin = kc * 32 + K;
  int c    = klin & 63;
  int khkw = klin >> 6;                          // 0..8
  int kh   = khkw / 3, kw = khkw - 3 * kh;
  int m    = mt * 16 + (lane & 15);
  float v  = w[(c * 9 + kh * 3 + kw) * COUT + m];
  __bf16 h = (__bf16)v;
  Ahi[f] = h;
  Alo[f] = (__bf16)(v - (float)h);
}

// ---------------------------------------------------------------------------
// Implicit-GEMM conv: block = 8 waves computes O=128 x N=128.
// Double-buffered LDS B staging: one barrier per K-chunk; A tile and gather
// loads register-double-buffered one chunk ahead (hidden behind 24 WMMAs).
// ---------------------------------------------------------------------------
__global__ __launch_bounds__(256) void conv_wmma(
    const uint32_t* __restrict__ xp,
    const __bf16*   __restrict__ Ahi,
    const __bf16*   __restrict__ Alo,
    float*          __restrict__ out) {
  // B staging: [s(8)][half(2)][n16(16)][e(16)] bf16, hi/lo planes, x2 buffers
  __shared__ __align__(32) __bf16 sBh[2][4096];
  __shared__ __align__(32) __bf16 sBl[2][4096];

  const int tid  = threadIdx.x;
  const int lane = tid & 31;
  const int wv   = tid >> 5;             // wave id == M tile
  const int n0   = blockIdx.x * NBLK;

  // gather geometry: thread -> (n = n0 + j, K in [g*16, g*16+16))
  const int j  = tid & 127;
  const int g  = tid >> 7;               // 0..1 == K half
  const int n  = n0 + j;
  const int b  = n / HWHW;  const int r0 = n - b * HWHW;
  const int ho = r0 / HW;   const int wo = r0 - ho * HW;
  const int pbase = (b * CIN + g * 16) * HPHP + ho * HP + wo;
  const int stoff = (j >> 4) * 512 + (g << 8) + (j & 15) * 16;

  const int aBase = (wv * 32 + lane) * 16;
  const int bRd   = ((lane >> 4) << 8) + ((lane & 15) << 4);

  v8f acc[8] = {};
  uint32_t pv[16];
  v16bf aH, aL, aHn, aLn;

  // chunk-uniform gather offset: (kc&1)*32*HPHP + kh*HP + kw
  auto chunk_off = [](int kc) {
    int pr = kc >> 1;
    int kh = pr / 3, kw = pr - 3 * kh;
    return (kc & 1) * (32 * HPHP) + kh * HP + kw;
  };

  // prologue: gather + A tile for chunk 0
  {
    const uint32_t* src = xp + pbase + chunk_off(0);
#pragma unroll
    for (int i = 0; i < 16; ++i) pv[i] = src[i * HPHP];
    aH = *(const v16bf*)(Ahi + aBase);
    aL = *(const v16bf*)(Alo + aBase);
    aHn = aH; aLn = aL;
  }

#pragma unroll 2
  for (int kc = 0; kc < KCH; ++kc) {
    __bf16* bh = sBh[kc & 1];
    __bf16* bl = sBl[kc & 1];
    // ---- unpack hi/lo, stage to LDS (4 x ds_store_b128) ----
#pragma unroll
    for (int q = 0; q < 2; ++q) {
      uint4 hp, lp;
      hp.x = (pv[8*q+0] & 0xffffu) | (pv[8*q+1] << 16);
      hp.y = (pv[8*q+2] & 0xffffu) | (pv[8*q+3] << 16);
      hp.z = (pv[8*q+4] & 0xffffu) | (pv[8*q+5] << 16);
      hp.w = (pv[8*q+6] & 0xffffu) | (pv[8*q+7] << 16);
      lp.x = (pv[8*q+0] >> 16) | (pv[8*q+1] & 0xffff0000u);
      lp.y = (pv[8*q+2] >> 16) | (pv[8*q+3] & 0xffff0000u);
      lp.z = (pv[8*q+4] >> 16) | (pv[8*q+5] & 0xffff0000u);
      lp.w = (pv[8*q+6] >> 16) | (pv[8*q+7] & 0xffff0000u);
      *(uint4*)(bh + stoff + 8*q) = hp;
      *(uint4*)(bl + stoff + 8*q) = lp;
    }
    __syncthreads();

    // ---- prefetch next chunk's gather + A tile (hidden behind WMMA) ----
    if (kc + 1 < KCH) {
      const uint32_t* src = xp + pbase + chunk_off(kc + 1);
#pragma unroll
      for (int i = 0; i < 16; ++i) pv[i] = src[i * HPHP];
      aHn = *(const v16bf*)(Ahi + (kc + 1) * APLANE + aBase);
      aLn = *(const v16bf*)(Alo + (kc + 1) * APLANE + aBase);
    }

    // ---- 8 N-subtiles x (hi*hi + lo*hi + hi*lo) ----
#pragma unroll
    for (int s = 0; s < 8; ++s) {
      v16bf b_hi = *(const v16bf*)&bh[s * 512 + bRd];
      v16bf b_lo = *(const v16bf*)&bl[s * 512 + bRd];
      acc[s] = __builtin_amdgcn_wmma_f32_16x16x32_bf16(
          false, aH, false, b_hi, (short)0, acc[s], false, false);
      acc[s] = __builtin_amdgcn_wmma_f32_16x16x32_bf16(
          false, aL, false, b_hi, (short)0, acc[s], false, false);
      acc[s] = __builtin_amdgcn_wmma_f32_16x16x32_bf16(
          false, aH, false, b_lo, (short)0, acc[s], false, false);
    }
    aH = aHn; aL = aLn;
  }

  // ---- store: C/D layout: elem r -> M = r + 8*(lane/16), N = lane%16 ----
  const int n16 = lane & 15;
  const int hlf = lane >> 4;
#pragma unroll
  for (int s = 0; s < 8; ++s) {
    int nn = n0 + s * 16 + n16;
    int bb = nn / HWHW;  int r2 = nn - bb * HWHW;
    int oh = r2 / HW;    int ow = r2 - oh * HW;
    float* op = out + (size_t)bb * (COUT * HWHW)
                    + (size_t)(wv * 16 + 8 * hlf) * HWHW + oh * HW + ow;
#pragma unroll
    for (int rr = 0; rr < 8; ++rr)
      op[(size_t)rr * HWHW] = acc[s][rr];
  }
}

extern "C" void kernel_launch(void* const* d_in, const int* in_sizes, int n_in,
                              void* d_out, int out_size, void* d_ws, size_t ws_size,
                              hipStream_t stream) {
  const float* x = (const float*)d_in[0];   // (16,64,112,112) f32
  const float* w = (const float*)d_in[1];   // (64,3,3,128) f32
  float* out = (float*)d_out;               // (16,128,112,112) f32

  // workspace layout (needs ~53.6 MB):
  uint32_t* xp  = (uint32_t*)d_ws;                  // XP_ELEMS dwords
  __bf16*   Ahi = (__bf16*)(xp + XP_ELEMS);         // 73728 bf16
  __bf16*   Alo = Ahi + A_ELEMS;                    // 73728 bf16

  prep_x<<<XP_ELEMS / 256, 256, 0, stream>>>(x, xp);
  prep_wswz<<<A_ELEMS / 256, 256, 0, stream>>>(w, Ahi, Alo);
  conv_wmma<<<NTOT / NBLK, 256, 0, stream>>>(xp, Ahi, Alo, out);
}
